// GCGALDA_3959959847088
// MI455X (gfx1250) — matrix operand order repaired and checked by reading, not compiled, not run on hardware
//
#include <hip/hip_runtime.h>
#include <hip/hip_bf16.h>
#include <math.h>

typedef __attribute__((ext_vector_type(16))) _Float16 v16h;
typedef __attribute__((ext_vector_type(8)))  _Float16 v8h;
typedef __attribute__((ext_vector_type(8)))  float    v8f;

// Problem constants (match reference)
#define N_RNA 2048
#define N_DIS 512
#define NN    2560            // N_RNA + N_DIS
#define EE    81920           // E
#define ET    84480           // E + N (self loops)
#define HEADS 8
#define HDIM  64
#define NPAIR (N_RNA * N_DIS) // 1048576

// ---------------------------------------------------------------------------
// Utility kernels
// ---------------------------------------------------------------------------
__global__ void fill_f32(float* __restrict__ p, float v, int n) {
  int i = blockIdx.x * 256 + threadIdx.x;
  if (i < n) p[i] = v;
}

// fp32 -> f16 row-major copy
__global__ void cvt_f16(const float* __restrict__ s, _Float16* __restrict__ d, int n) {
  int i = blockIdx.x * 256 + threadIdx.x;
  if (i < n) d[i] = (_Float16)s[i];
}

// fp32 [K,N] -> f16 transposed [N,K]:  d[n*K+k] = s[k*N+n]
__global__ void cvt_f16_T(const float* __restrict__ s, _Float16* __restrict__ d,
                          int K, int N) {
  int i = blockIdx.x * 256 + threadIdx.x;
  if (i >= K * N) return;
  int k = i / N, n = i % N;
  d[(size_t)n * K + k] = (_Float16)s[i];
}

__device__ inline float atomicMaxFloat(float* addr, float value) {
  if (value >= 0.0f)
    return __int_as_float(atomicMax((int*)addr, __float_as_int(value)));
  else
    return __uint_as_float(atomicMin((unsigned int*)addr, __float_as_uint(value)));
}

// ---------------------------------------------------------------------------
// WMMA GEMM on pre-converted f16:
//   C[M,N] (f32) = Ah[M,K] (f16 row-major) @ BhT (f16, [N,K] col-major of B)
// One wave per 16x16 C tile. Per K-step: 2x b128 loads (A), 2x b128 (B), 1 WMMA.
// Fragment layouts: contiguous K, elem s of lane half h = k 16h+s in A (row) and B (lane=col).
// ---------------------------------------------------------------------------
__global__ void wmma_gemm_f16(const _Float16* __restrict__ Ah,
                              const _Float16* __restrict__ BhT,
                              float* __restrict__ C, int M, int Nn, int K) {
  int wid  = (blockIdx.x * blockDim.x + threadIdx.x) >> 5;
  int lane = threadIdx.x & 31;
  int ntn  = Nn >> 4;
  int total = (M >> 4) * ntn;
  if (wid >= total) return;
  int tm = wid / ntn, tn = wid % ntn;
  int hi  = lane >> 4;   // lane half selects K sub-block
  int l15 = lane & 15;
  int arow = tm * 16 + l15;
  int bcol = tn * 16 + l15;
  const _Float16* Abase = Ah + (size_t)arow * K + hi * 16;
  const _Float16* Bbase = BhT + (size_t)bcol * K + hi * 16;
  v8f c = {};
  for (int k0 = 0; k0 < K; k0 += 32) {
    // A fragment: elems 0-15 = K kbase..+15 (16 consecutive K, matching B)
    v8h alo = *(const v8h*)(Abase + k0);
    v8h ahi = *(const v8h*)(Abase + k0 + 8);
    v16h a = __builtin_shufflevector(alo, ahi, 0, 1, 2, 3, 4, 5, 6, 7,
                                     8, 9, 10, 11, 12, 13, 14, 15);
    // B fragment: 16 consecutive K values for this column
    v16h b = *(const v16h*)(Bbase + k0);
    c = __builtin_amdgcn_wmma_f32_16x16x32_f16(false, a, false, b, (short)0, c,
                                               false, false);
  }
  // C layout: lane 0-15 -> N=lane, M=v ; lane 16-31 -> N=lane-16, M=8+v
  int rbase = tm * 16 + hi * 8;
#pragma unroll
  for (int v = 0; v < 8; ++v)
    C[(size_t)(rbase + v) * Nn + tn * 16 + l15] = c[v];
}

// ---------------------------------------------------------------------------
// GAT attention kernels
// ---------------------------------------------------------------------------
__global__ void attn_scores(const float* __restrict__ h, const float* __restrict__ asrc,
                            const float* __restrict__ adst, float* __restrict__ es,
                            float* __restrict__ ed) {
  int idx = blockIdx.x * 256 + threadIdx.x;       // n*HEADS + hh
  if (idx >= NN * HEADS) return;
  int n = idx >> 3, hh = idx & 7;
  const float* hp = h + n * (HEADS * HDIM) + hh * HDIM;
  float s = 0.f, d = 0.f;
  for (int t = 0; t < HDIM; ++t) {
    float v = hp[t];
    s += v * asrc[hh * HDIM + t];
    d += v * adst[hh * HDIM + t];
  }
  es[idx] = s;
  ed[idx] = d;
}

__device__ inline int edge_src(const int* ei, int e) { return e < EE ? ei[e] : e - EE; }
__device__ inline int edge_dst(const int* ei, int e) { return e < EE ? ei[EE + e] : e - EE; }

__global__ void edge_max(const int* __restrict__ ei, const float* __restrict__ es,
                         const float* __restrict__ ed, float* __restrict__ m) {
  int idx = blockIdx.x * 256 + threadIdx.x;       // e*HEADS + hh
  if (idx >= ET * HEADS) return;
  int e = idx >> 3, hh = idx & 7;
  int s = edge_src(ei, e), d = edge_dst(ei, e);
  float v = es[s * HEADS + hh] + ed[d * HEADS + hh];
  v = v > 0.f ? v : 0.2f * v;                     // leaky_relu 0.2
  atomicMaxFloat(&m[d * HEADS + hh], v);
}

__global__ void edge_exp(const int* __restrict__ ei, const float* __restrict__ es,
                         const float* __restrict__ ed, const float* __restrict__ m,
                         float* __restrict__ eedge, float* __restrict__ ssum) {
  int idx = blockIdx.x * 256 + threadIdx.x;
  if (idx >= ET * HEADS) return;
  int e = idx >> 3, hh = idx & 7;
  int s = edge_src(ei, e), d = edge_dst(ei, e);
  float v = es[s * HEADS + hh] + ed[d * HEADS + hh];
  v = v > 0.f ? v : 0.2f * v;
  float ex = __expf(v - m[d * HEADS + hh]);
  eedge[idx] = ex;
  atomicAdd(&ssum[d * HEADS + hh], ex);
}

__global__ void edge_agg_gat(const int* __restrict__ ei, const float* __restrict__ hlin,
                             const float* __restrict__ eedge, const float* __restrict__ ssum,
                             float* __restrict__ hagg) {
  int idx = blockIdx.x * 256 + threadIdx.x;       // e*512 + t  (43.25M, fits int)
  if (idx >= ET * 512) return;
  int e = idx >> 9, t = idx & 511, hh = t >> 6;
  int s = edge_src(ei, e), d = edge_dst(ei, e);
  float alpha = eedge[e * HEADS + hh] / (ssum[d * HEADS + hh] + 1e-16f);
  atomicAdd(&hagg[d * 512 + t], hlin[s * 512 + t] * alpha);
}

// bias + elu, writing both f32 (in place) and f16 copy for the next GEMM
__global__ void bias_elu_f16(float* __restrict__ x, const float* __restrict__ b,
                             _Float16* __restrict__ xh, int n, int cols) {
  int i = blockIdx.x * 256 + threadIdx.x;
  if (i >= n) return;
  float v = x[i] + b[i % cols];
  v = v > 0.f ? v : (__expf(v) - 1.f);
  x[i] = v;
  xh[i] = (_Float16)v;
}

__global__ void bias_relu_f16(float* __restrict__ x, const float* __restrict__ b,
                              _Float16* __restrict__ xh, int n, int cols) {
  int i = blockIdx.x * 256 + threadIdx.x;
  if (i >= n) return;
  float v = fmaxf(x[i] + b[i % cols], 0.f);
  x[i] = v;
  xh[i] = (_Float16)v;
}

// ---------------------------------------------------------------------------
// GCN kernels
// ---------------------------------------------------------------------------
__global__ void deg_kernel(const int* __restrict__ ei, float* __restrict__ deg) {
  int e = blockIdx.x * 256 + threadIdx.x;
  if (e >= ET) return;
  atomicAdd(&deg[edge_dst(ei, e)], 1.f);
}

__global__ void dinv_kernel(const float* __restrict__ deg, float* __restrict__ dinv) {
  int i = blockIdx.x * 256 + threadIdx.x;
  if (i >= NN) return;
  float g = deg[i];
  dinv[i] = g > 0.f ? rsqrtf(fmaxf(g, 1e-12f)) : 0.f;
}

__global__ void gcn_agg(const int* __restrict__ ei, const float* __restrict__ xw,
                        const float* __restrict__ dinv, float* __restrict__ out, int cols) {
  int idx = blockIdx.x * 256 + threadIdx.x;
  if (idx >= ET * cols) return;
  int e = idx / cols, t = idx % cols;
  int s = edge_src(ei, e), d = edge_dst(ei, e);
  float norm = dinv[s] * dinv[d];
  atomicAdd(&out[d * cols + t], xw[s * cols + t] * norm);
}

// ---------------------------------------------------------------------------
// Fused pair MLP: pred[i,j] = sigmoid( leaky(leaky(A[i]+B[j]+b1) @ W2 + b2) @ w3 )
// One block (256 thr = 8 waves) per RNA node i; loops over all 32 j-tiles,
// reusing A[i]+b1. Wave w owns 16-column tile w of W2 (via pre-transposed f16
// W2T[128,64] read straight from L2 with v16h loads). z1 staged as f16 in LDS.
// ---------------------------------------------------------------------------
__global__ void pair_mlp(const float* __restrict__ Ap, const float* __restrict__ Bp,
                         const float* __restrict__ b1,
                         const _Float16* __restrict__ W2T,   // [128,64] = W2 transposed
                         const float* __restrict__ b2, const float* __restrict__ w3,
                         float* __restrict__ pred) {
  __shared__ __align__(32) _Float16 z1h[16 * 64];
  __shared__ float arow[64];
  __shared__ float acc[16];
  int tid = threadIdx.x;
  int i = blockIdx.x;

  if (tid < 64) arow[tid] = Ap[i * 64 + tid] + b1[tid];

  int wave = tid >> 5, lane = tid & 31;
  int hi = lane >> 4, l15 = lane & 15;
  int col = wave * 16 + l15;
  // B fragments are loop-invariant: hoist them (W2T row = col, 64 K-values)
  const _Float16* wb = W2T + (size_t)col * 64 + hi * 16;
  v16h b0 = *(const v16h*)(wb);        // K 0..31 slice for this lane-half
  v16h b1f = *(const v16h*)(wb + 32);  // K 32..63 slice
  float b2c = b2[col], w3c = w3[col];

  for (int jt = 0; jt < 32; ++jt) {
    int j0 = jt << 4;
    __syncthreads();                   // protect z1h/acc from prior iteration
    if (tid < 16) acc[tid] = 0.f;
    for (int idx = tid; idx < 16 * 64; idx += 256) {
      int r = idx >> 6, k = idx & 63;
      float v = arow[k] + Bp[(j0 + r) * 64 + k];
      z1h[idx] = (_Float16)(v > 0.f ? v : 0.01f * v);   // leaky 0.01
    }
    __syncthreads();

    v8f c = {};
    const _Float16* za = &z1h[l15 * 64 + hi * 16];
    {
      v8h alo = *(const v8h*)(za);
      v8h ahi = *(const v8h*)(za + 8);
      v16h a = __builtin_shufflevector(alo, ahi, 0, 1, 2, 3, 4, 5, 6, 7,
                                       8, 9, 10, 11, 12, 13, 14, 15);
      c = __builtin_amdgcn_wmma_f32_16x16x32_f16(false, a, false, b0, (short)0, c,
                                                 false, false);
    }
    {
      v8h alo = *(const v8h*)(za + 32);
      v8h ahi = *(const v8h*)(za + 40);
      v16h a = __builtin_shufflevector(alo, ahi, 0, 1, 2, 3, 4, 5, 6, 7,
                                       8, 9, 10, 11, 12, 13, 14, 15);
      c = __builtin_amdgcn_wmma_f32_16x16x32_f16(false, a, false, b1f, (short)0, c,
                                                 false, false);
    }
#pragma unroll
    for (int v = 0; v < 8; ++v) {
      int row = hi * 8 + v;
      float z = c[v] + b2c;
      z = z > 0.f ? z : 0.01f * z;                      // leaky 0.01
      atomicAdd(&acc[row], z * w3c);                    // ds_add_f32
    }
    __syncthreads();
    if (tid < 16) {
      float s = acc[tid];
      pred[i * N_DIS + j0 + tid] = 1.f / (1.f + __expf(-s));
    }
  }
}

__global__ void copy_labels(const float* __restrict__ rel, float* __restrict__ out) {
  int i = blockIdx.x * 256 + threadIdx.x;
  if (i < NPAIR) out[i] = rel[i];
}

// ---------------------------------------------------------------------------
// Host launch
// ---------------------------------------------------------------------------
static inline int gb(long long n) { return (int)((n + 255) / 256); }

extern "C" void kernel_launch(void* const* d_in, const int* in_sizes, int n_in,
                              void* d_out, int out_size, void* d_ws, size_t ws_size,
                              hipStream_t stream) {
  const float* x       = (const float*)d_in[0];
  const int*   ei      = (const int*)d_in[1];
  const float* rel     = (const float*)d_in[2];
  // d_in[3] = train_model (unused)
  const float* W_gat   = (const float*)d_in[4];
  const float* att_src = (const float*)d_in[5];
  const float* att_dst = (const float*)d_in[6];
  const float* b_gat   = (const float*)d_in[7];
  const float* W_gcn1  = (const float*)d_in[8];
  const float* b_gcn1  = (const float*)d_in[9];
  const float* W_gcn2  = (const float*)d_in[10];
  const float* b_gcn2  = (const float*)d_in[11];
  const float* W_mlp1  = (const float*)d_in[12];
  const float* b_mlp1  = (const float*)d_in[13];
  const float* W_mlp2  = (const float*)d_in[14];
  const float* b_mlp2  = (const float*)d_in[15];
  const float* W_mlp3  = (const float*)d_in[16];

  // --- f32 workspace region ---
  float* W = (float*)d_ws;
  size_t o = 0;
  float* h_lin = W + o; o += (size_t)NN * 512;
  float* h_agg = W + o; o += (size_t)NN * 512;   // becomes h_elu in place
  float* es_   = W + o; o += NN * HEADS;
  float* ed_   = W + o; o += NN * HEADS;
  float* m_    = W + o; o += NN * HEADS;
  float* s_    = W + o; o += NN * HEADS;
  float* eedge = W + o; o += (size_t)ET * HEADS;
  float* deg   = W + o; o += NN;
  float* dinv  = W + o; o += NN;
  float* xw1   = W + o; o += (size_t)NN * 256;
  float* g1    = W + o; o += (size_t)NN * 256;
  float* xw2   = W + o; o += (size_t)NN * 128;
  float* g2    = W + o; o += (size_t)NN * 128;
  float* Apart = W + o; o += (size_t)N_RNA * 64;
  float* Bpart = W + o; o += (size_t)N_DIS * 64;

  // --- f16 workspace region (starts 256B-aligned: o is a multiple of 64) ---
  _Float16* H = (_Float16*)(W + o);
  size_t q = 0;
  _Float16* xh    = H + q; q += (size_t)NN * 512;    // x, f16
  _Float16* WgatT = H + q; q += (size_t)512 * 512;   // W_gat^T
  _Float16* h16   = H + q; q += (size_t)NN * 512;    // elu(h) f16
  _Float16* Wg1T  = H + q; q += (size_t)256 * 512;   // W_gcn1^T
  _Float16* g1h   = H + q; q += (size_t)NN * 256;
  _Float16* Wg2T  = H + q; q += (size_t)128 * 256;   // W_gcn2^T
  _Float16* g2h   = H + q; q += (size_t)NN * 128;
  _Float16* W1aT  = H + q; q += (size_t)64 * 128;    // W_mlp1[:128]^T
  _Float16* W1bT  = H + q; q += (size_t)64 * 128;    // W_mlp1[128:]^T
  _Float16* W2T   = H + q; q += (size_t)128 * 64;    // W_mlp2^T

  float* pred   = (float*)d_out;
  float* labels = pred + NPAIR;

  // ---- one-time f16 conversions / weight transposes ----
  cvt_f16<<<gb((long long)NN * 512), 256, 0, stream>>>(x, xh, NN * 512);
  cvt_f16_T<<<gb(512 * 512), 256, 0, stream>>>(W_gat, WgatT, 512, 512);
  cvt_f16_T<<<gb(512 * 256), 256, 0, stream>>>(W_gcn1, Wg1T, 512, 256);
  cvt_f16_T<<<gb(256 * 128), 256, 0, stream>>>(W_gcn2, Wg2T, 256, 128);
  cvt_f16_T<<<gb(128 * 64), 256, 0, stream>>>(W_mlp1, W1aT, 128, 64);
  cvt_f16_T<<<gb(128 * 64), 256, 0, stream>>>(W_mlp1 + 128 * 64, W1bT, 128, 64);
  cvt_f16_T<<<gb(64 * 128), 256, 0, stream>>>(W_mlp2, W2T, 64, 128);

  // ---- GAT ----
  wmma_gemm_f16<<<640, 256, 0, stream>>>(xh, WgatT, h_lin, NN, 512, 512);
  attn_scores<<<gb(NN * HEADS), 256, 0, stream>>>(h_lin, att_src, att_dst, es_, ed_);
  fill_f32<<<gb(NN * HEADS), 256, 0, stream>>>(m_, -3.0e38f, NN * HEADS);
  fill_f32<<<gb(NN * HEADS), 256, 0, stream>>>(s_, 0.f, NN * HEADS);
  fill_f32<<<gb((long long)NN * 512), 256, 0, stream>>>(h_agg, 0.f, NN * 512);
  edge_max<<<gb((long long)ET * HEADS), 256, 0, stream>>>(ei, es_, ed_, m_);
  edge_exp<<<gb((long long)ET * HEADS), 256, 0, stream>>>(ei, es_, ed_, m_, eedge, s_);
  edge_agg_gat<<<gb((long long)ET * 512), 256, 0, stream>>>(ei, h_lin, eedge, s_, h_agg);
  bias_elu_f16<<<gb((long long)NN * 512), 256, 0, stream>>>(h_agg, b_gat, h16,
                                                            NN * 512, 512);

  // ---- GCN degree norm ----
  fill_f32<<<gb(NN), 256, 0, stream>>>(deg, 0.f, NN);
  deg_kernel<<<gb(ET), 256, 0, stream>>>(ei, deg);
  dinv_kernel<<<gb(NN), 256, 0, stream>>>(deg, dinv);

  // ---- GCN1 ----
  wmma_gemm_f16<<<320, 256, 0, stream>>>(h16, Wg1T, xw1, NN, 256, 512);
  fill_f32<<<gb((long long)NN * 256), 256, 0, stream>>>(g1, 0.f, NN * 256);
  gcn_agg<<<gb((long long)ET * 256), 256, 0, stream>>>(ei, xw1, dinv, g1, 256);
  bias_elu_f16<<<gb((long long)NN * 256), 256, 0, stream>>>(g1, b_gcn1, g1h,
                                                            NN * 256, 256);

  // ---- GCN2 ----
  wmma_gemm_f16<<<160, 256, 0, stream>>>(g1h, Wg2T, xw2, NN, 128, 256);
  fill_f32<<<gb((long long)NN * 128), 256, 0, stream>>>(g2, 0.f, NN * 128);
  gcn_agg<<<gb((long long)ET * 128), 256, 0, stream>>>(ei, xw2, dinv, g2, 128);
  // relu(elu(y)) == relu(y): apply bias+relu directly -> "outputs"
  bias_relu_f16<<<gb((long long)NN * 128), 256, 0, stream>>>(g2, b_gcn2, g2h,
                                                             NN * 128, 128);

  // ---- MLP layer-1 factorization: A = r @ W1[:128], B = d @ W1[128:] ----
  wmma_gemm_f16<<<64, 256, 0, stream>>>(g2h, W1aT, Apart, N_RNA, 64, 128);
  wmma_gemm_f16<<<16, 256, 0, stream>>>(g2h + (size_t)N_RNA * 128, W1bT,
                                        Bpart, N_DIS, 64, 128);

  // ---- Fused pair MLP (dominant compute, WMMA) ----
  pair_mlp<<<N_RNA, 256, 0, stream>>>(Apart, Bpart, b_mlp1, W2T, b_mlp2, W_mlp3, pred);

  // ---- labels ----
  copy_labels<<<gb(NPAIR), 256, 0, stream>>>(rel, labels);

  (void)in_sizes; (void)n_in; (void)out_size; (void)ws_size;
}
